// ViT_68298569941454
// MI455X (gfx1250) — matrix operand order repaired and checked by reading, not compiled, not run on hardware
//
#include <hip/hip_runtime.h>
#include <math.h>

typedef __attribute__((ext_vector_type(16))) _Float16 v16h;
typedef __attribute__((ext_vector_type(8)))  float    v8f;

namespace vit {
constexpr int B = 64, C = 3, IMG = 224, P = 16, D = 384, H = 6, L = 6, F = 1536, NC = 1000;
constexpr int GP = IMG / P;        // 14
constexpr int NP = GP * GP;        // 196
constexpr int S  = NP + 1;         // 197
constexpr int PD = C * P * P;      // 768
constexpr int DH = D / H;          // 64
constexpr int SP = 224;            // padded S (multiple of 32) for P@V K-dim
constexpr int NCP = 1008;          // padded NC
constexpr int BS  = B * S;         // 12608
constexpr int BNP = B * NP;        // 12544
}

union FragAB { v16h v; uint4 q[2]; };
union FragC  { v8f  v; float f[8]; };

__device__ __forceinline__ uint4 zero4() { uint4 z; z.x = z.y = z.z = z.w = 0u; return z; }

enum GemmMode { M_F32 = 0, M_GELU_F16 = 1, M_RES_F32 = 2, M_F16 = 3, M_SCALE_F32 = 4, M_QKH = 5, M_VT = 6 };

// Load one pipeline stage: 2 A-frags (rows) + 4 B-frags (cols), 12 x b128.
#define LOAD_SET(a0, a1, b0, b1, b2, b3, kk)                                   \
  do {                                                                         \
    const uint4* p0_ = reinterpret_cast<const uint4*>(Ar0 + (kk));             \
    const uint4* p1_ = reinterpret_cast<const uint4*>(Ar1 + (kk));             \
    (a0).q[0] = p0_[0]; (a0).q[1] = p0_[2];                                    \
    (a1).q[0] = p1_[0]; (a1).q[1] = p1_[2];                                    \
    const uint4* q0_ = reinterpret_cast<const uint4*>(Wr0 + (kk));             \
    const uint4* q1_ = reinterpret_cast<const uint4*>(Wr1 + (kk));             \
    const uint4* q2_ = reinterpret_cast<const uint4*>(Wr2 + (kk));             \
    const uint4* q3_ = reinterpret_cast<const uint4*>(Wr3 + (kk));             \
    (b0).q[0] = q0_[0]; (b0).q[1] = q0_[2];                                    \
    (b1).q[0] = q1_[0]; (b1).q[1] = q1_[2];                                    \
    (b2).q[0] = q2_[0]; (b2).q[1] = q2_[2];                                    \
    (b3).q[0] = q3_[0]; (b3).q[1] = q3_[2];                                    \
  } while (0)

// Consume one stage: 8 WMMAs (2 row-tiles x 4 col-tiles).
#define MMA_SET(a0, a1, b0, b1, b2, b3)                                        \
  do {                                                                         \
    acc[0][0].v = __builtin_amdgcn_wmma_f32_16x16x32_f16(                      \
        false, (a0).v, false, (b0).v, (short)0, acc[0][0].v, false, false);    \
    acc[1][0].v = __builtin_amdgcn_wmma_f32_16x16x32_f16(                      \
        false, (a1).v, false, (b0).v, (short)0, acc[1][0].v, false, false);    \
    acc[0][1].v = __builtin_amdgcn_wmma_f32_16x16x32_f16(                      \
        false, (a0).v, false, (b1).v, (short)0, acc[0][1].v, false, false);    \
    acc[1][1].v = __builtin_amdgcn_wmma_f32_16x16x32_f16(                      \
        false, (a1).v, false, (b1).v, (short)0, acc[1][1].v, false, false);    \
    acc[0][2].v = __builtin_amdgcn_wmma_f32_16x16x32_f16(                      \
        false, (a0).v, false, (b2).v, (short)0, acc[0][2].v, false, false);    \
    acc[1][2].v = __builtin_amdgcn_wmma_f32_16x16x32_f16(                      \
        false, (a1).v, false, (b2).v, (short)0, acc[1][2].v, false, false);    \
    acc[0][3].v = __builtin_amdgcn_wmma_f32_16x16x32_f16(                      \
        false, (a0).v, false, (b3).v, (short)0, acc[0][3].v, false, false);    \
    acc[1][3].v = __builtin_amdgcn_wmma_f32_16x16x32_f16(                      \
        false, (a1).v, false, (b3).v, (short)0, acc[1][3].v, false, false);    \
  } while (0)

// Generic batched WMMA GEMM: out = epilogue(A[MxK,f16] * Wt[NxK,f16]^T (+bias))
// Wt holds B transposed: row n = column n of the K x N weight, contiguous in K.
// One wave -> 32(M) x 64(N) strip; fast path is 2-stage software-pipelined.
template <int MODE>
__global__ __launch_bounds__(256) void gemm_wmma(
    const _Float16* __restrict__ A, long long aB, int lda,
    const _Float16* __restrict__ W, long long wB, int ldw,
    const float* __restrict__ bias,
    void* __restrict__ out, long long oB, int ldo,
    const float* __restrict__ res,
    int M, int N, int K, float scale)
{
  using namespace vit;
  const int z    = blockIdx.z;
  const int wave = threadIdx.x >> 5;
  const int lane = threadIdx.x & 31;
  const int hid  = lane >> 4;     // half-wave id
  const int l16  = lane & 15;
  const int tileM = blockIdx.y * 8 + wave;
  const int row0  = tileM << 5;   // 32 rows per wave
  if (row0 >= M) return;
  const int nt0 = blockIdx.x << 2;

  const _Float16* Ab = A + (long long)z * aB;
  const _Float16* Wb = W + (long long)z * wB;

  FragC acc[2][4];
#pragma unroll
  for (int i = 0; i < 2; ++i)
#pragma unroll
    for (int j = 0; j < 4; ++j)
#pragma unroll
      for (int e = 0; e < 8; ++e) acc[i][j].f[e] = 0.0f;

  const bool full = (row0 + 32 <= M) && (((nt0 << 4) + 64) <= N);

  if (full) {
    // ---------- fast path: unconditional loads, 2-stage ping-pong pipeline ----------
    const _Float16* Ar0 = Ab + (long long)(row0 + l16) * lda + hid * 8;
    const _Float16* Ar1 = Ar0 + (long long)16 * lda;
    const _Float16* Wr0 = Wb + (long long)(((nt0 + 0) << 4) + l16) * ldw + hid * 8;
    const _Float16* Wr1 = Wb + (long long)(((nt0 + 1) << 4) + l16) * ldw + hid * 8;
    const _Float16* Wr2 = Wb + (long long)(((nt0 + 2) << 4) + l16) * ldw + hid * 8;
    const _Float16* Wr3 = Wb + (long long)(((nt0 + 3) << 4) + l16) * ldw + hid * 8;

    FragAB xa0, xa1, xb0, xb1, xb2, xb3;   // stage X
    FragAB ya0, ya1, yb0, yb1, yb2, yb3;   // stage Y

    const int steps = K >> 5;              // >= 2 for all our GEMMs
    LOAD_SET(xa0, xa1, xb0, xb1, xb2, xb3, 0);
    int step = 1;
    for (; step + 1 < steps; step += 2) {
      LOAD_SET(ya0, ya1, yb0, yb1, yb2, yb3, step << 5);
      MMA_SET(xa0, xa1, xb0, xb1, xb2, xb3);
      LOAD_SET(xa0, xa1, xb0, xb1, xb2, xb3, (step + 1) << 5);
      MMA_SET(ya0, ya1, yb0, yb1, yb2, yb3);
    }
    if (step < steps) {
      LOAD_SET(ya0, ya1, yb0, yb1, yb2, yb3, step << 5);
      MMA_SET(xa0, xa1, xb0, xb1, xb2, xb3);
      MMA_SET(ya0, ya1, yb0, yb1, yb2, yb3);
    } else {
      MMA_SET(xa0, xa1, xb0, xb1, xb2, xb3);
    }
  } else {
    // ---------- guarded path: edge tiles only ----------
    const int  am0 = row0 + l16;
    const int  am1 = row0 + 16 + l16;
    const bool av0 = (am0 < M), av1 = (am1 < M);
    const _Float16* Ar0 = Ab + (long long)(av0 ? am0 : 0) * lda + hid * 8;
    const _Float16* Ar1 = Ab + (long long)(av1 ? am1 : 0) * lda + hid * 8;
    for (int kk = 0; kk < K; kk += 32) {
      FragAB a0, a1;
      if (av0) {
        const uint4* p = reinterpret_cast<const uint4*>(Ar0 + kk);
        a0.q[0] = p[0]; a0.q[1] = p[2];
      } else { a0.q[0] = zero4(); a0.q[1] = zero4(); }
      if (av1) {
        const uint4* p = reinterpret_cast<const uint4*>(Ar1 + kk);
        a1.q[0] = p[0]; a1.q[1] = p[2];
      } else { a1.q[0] = zero4(); a1.q[1] = zero4(); }
#pragma unroll
      for (int j = 0; j < 4; ++j) {
        const int bn = ((nt0 + j) << 4) + l16;
        FragAB b;
        if (bn < N) {
          const uint4* p = reinterpret_cast<const uint4*>(Wb + (long long)bn * ldw + kk + hid * 8);
          b.q[0] = p[0]; b.q[1] = p[2];
        } else { b.q[0] = zero4(); b.q[1] = zero4(); }
        acc[0][j].v = __builtin_amdgcn_wmma_f32_16x16x32_f16(
            false, a0.v, false, b.v, (short)0, acc[0][j].v, false, false);
        acc[1][j].v = __builtin_amdgcn_wmma_f32_16x16x32_f16(
            false, a1.v, false, b.v, (short)0, acc[1][j].v, false, false);
      }
    }
  }

  // C/D layout: VGPR r -> row = rowbase + hid*8 + r, col = tile*16 + l16
#pragma unroll
  for (int i = 0; i < 2; ++i) {
#pragma unroll
    for (int j = 0; j < 4; ++j) {
      const int col = ((nt0 + j) << 4) + l16;
      if (col >= N) continue;
#pragma unroll
      for (int r = 0; r < 8; ++r) {
        const int row = row0 + i * 16 + hid * 8 + r;
        if (row >= M) continue;
        float v = acc[i][j].f[r];
        if constexpr (MODE == M_F32) {
          if (bias) v += bias[col];
          ((float*)out)[(long long)z * oB + (long long)row * ldo + col] = v;
        } else if constexpr (MODE == M_GELU_F16) {
          v += bias[col];
          v = 0.5f * v * (1.0f + erff(v * 0.70710678118654752f));
          ((_Float16*)out)[(long long)row * ldo + col] = (_Float16)v;
        } else if constexpr (MODE == M_RES_F32) {
          v += bias[col] + res[(long long)row * ldo + col];
          ((float*)out)[(long long)row * ldo + col] = v;
        } else if constexpr (MODE == M_F16) {
          ((_Float16*)out)[(long long)z * oB + (long long)row * ldo + col] = (_Float16)v;
        } else if constexpr (MODE == M_SCALE_F32) {
          v = (v + bias[col]) * scale;
          ((float*)out)[(long long)row * ldo + col] = v;
        } else if constexpr (MODE == M_QKH) {
          // (B*S, D) -> (B,H,S,DH) f16
          const int b_ = row / S, s_ = row % S, h_ = col / DH, d_ = col % DH;
          ((_Float16*)out)[((long long)(b_ * H + h_) * S + s_) * DH + d_] = (_Float16)v;
        } else { // M_VT: (B*S, D) -> (B,H,DH,SP) f16 (transposed per head, padded S)
          const int b_ = row / S, s_ = row % S, h_ = col / DH, d_ = col % DH;
          ((_Float16*)out)[((long long)(b_ * H + h_) * DH + d_) * SP + s_] = (_Float16)v;
        }
      }
    }
  }
}

// img (B,C,224,224) f32 -> patch matrix (B*NP, PD) f16, PD index = (py*P+px)*C + c
__global__ void patchify(const float* __restrict__ img, _Float16* __restrict__ out, long long n)
{
  using namespace vit;
  long long i = (long long)blockIdx.x * 256 + threadIdx.x;
  if (i >= n) return;
  int col = (int)(i % PD); long long pr = i / PD;
  int b = (int)(pr / NP), p = (int)(pr % NP);
  int gy = p / GP, gx = p % GP;
  int c = col % C, t = col / C;
  int py = t / P, px = t % P;
  long long idx = (((long long)(b * C + c) * IMG) + gy * P + py) * IMG + gx * P + px;
  out[i] = (_Float16)img[idx];
}

// W (K x N) f32 row-major -> out (Npad rows x K) f16, zero-padded rows
__global__ void wt_convert(const float* __restrict__ W, _Float16* __restrict__ out,
                           int K, int N, long long n)
{
  long long i = (long long)blockIdx.x * 256 + threadIdx.x;
  if (i >= n) return;
  int nn = (int)(i / K), k = (int)(i % K);
  out[i] = (nn < N) ? (_Float16)W[(long long)k * N + nn] : (_Float16)0.0f;
}

__global__ void zero_f16(_Float16* __restrict__ p, long long n)
{
  long long i = (long long)blockIdx.x * 256 + threadIdx.x;
  if (i < n) p[i] = (_Float16)0.0f;
}

__global__ void add_cls_pos(const float* __restrict__ pe, const float* __restrict__ cls,
                            const float* __restrict__ pos, float* __restrict__ X, long long n)
{
  using namespace vit;
  long long i = (long long)blockIdx.x * 256 + threadIdx.x;
  if (i >= n) return;
  int d = (int)(i % D); long long r = i / D;
  int b = (int)(r / S), s = (int)(r % S);
  float v = (s == 0) ? cls[d] : pe[((long long)b * NP + (s - 1)) * D + d];
  X[i] = v + pos[(long long)s * D + d];
}

__global__ void add_ctx(float* __restrict__ X, const float* __restrict__ ctx, long long n)
{
  using namespace vit;
  long long i = (long long)blockIdx.x * 256 + threadIdx.x;
  if (i >= n) return;
  int d = (int)(i % D); long long r = i / D;
  int b = (int)(r / S), s = (int)(r % S);
  int h = d / DH, dd = d % DH;
  X[i] += ctx[((long long)(b * H + h) * S + s) * DH + dd];
}

__global__ __launch_bounds__(128) void layernorm_f16(
    const float* __restrict__ X, long long rstride,
    const float* __restrict__ g, const float* __restrict__ bta,
    _Float16* __restrict__ out, long long ostride)
{
  using namespace vit;
  __shared__ float red[128];
  const float* x = X + (long long)blockIdx.x * rstride;
  int t = threadIdx.x;
  float s = 0.f, s2 = 0.f;
  for (int i = t; i < D; i += 128) { float v = x[i]; s += v; s2 += v * v; }
  red[t] = s; __syncthreads();
  for (int o = 64; o > 0; o >>= 1) { if (t < o) red[t] += red[t + o]; __syncthreads(); }
  float mean = red[0] * (1.0f / D); __syncthreads();
  red[t] = s2; __syncthreads();
  for (int o = 64; o > 0; o >>= 1) { if (t < o) red[t] += red[t + o]; __syncthreads(); }
  float var = red[0] * (1.0f / D) - mean * mean;
  float inv = rsqrtf(var + 1e-5f);
  _Float16* o_ = out + (long long)blockIdx.x * ostride;
  for (int i = t; i < D; i += 128) o_[i] = (_Float16)((x[i] - mean) * inv * g[i] + bta[i]);
}

// one block per attention row; reads logits f16 * scl, writes probs f16 incl. zero pad cols [S,SP)
__global__ __launch_bounds__(256) void softmax_rows(_Float16* __restrict__ Pr, float scl)
{
  using namespace vit;
  __shared__ float red[256];
  long long base = (long long)blockIdx.x * SP;
  int t = threadIdx.x;
  float v = (t < S) ? (float)Pr[base + t] * scl : -3.0e38f;
  red[t] = v; __syncthreads();
  for (int o = 128; o > 0; o >>= 1) { if (t < o) red[t] = fmaxf(red[t], red[t + o]); __syncthreads(); }
  float mx = red[0]; __syncthreads();
  float e = (t < S) ? __expf(v - mx) : 0.0f;
  red[t] = e; __syncthreads();
  for (int o = 128; o > 0; o >>= 1) { if (t < o) red[t] += red[t + o]; __syncthreads(); }
  float inv = 1.0f / red[0];
  if (t < SP) Pr[base + t] = (_Float16)(e * inv);
}

extern "C" void kernel_launch(void* const* d_in, const int* in_sizes, int n_in,
                              void* d_out, int out_size, void* d_ws, size_t ws_size,
                              hipStream_t stream)
{
  using namespace vit;
  const float* img    = (const float*)d_in[0];
  const float* patchw = (const float*)d_in[1];
  const float* patchb = (const float*)d_in[2];
  const float* cls    = (const float*)d_in[3];
  const float* pos    = (const float*)d_in[4];
  const float* ln1g   = (const float*)d_in[5];
  const float* ln1b   = (const float*)d_in[6];
  const float* wq     = (const float*)d_in[7];
  const float* wk     = (const float*)d_in[8];
  const float* wv     = (const float*)d_in[9];
  const float* ln2g   = (const float*)d_in[10];
  const float* ln2b   = (const float*)d_in[11];
  const float* w1     = (const float*)d_in[12];
  const float* b1     = (const float*)d_in[13];
  const float* w2     = (const float*)d_in[14];
  const float* b2     = (const float*)d_in[15];
  const float* hg     = (const float*)d_in[16];
  const float* hb     = (const float*)d_in[17];
  const float* hw     = (const float*)d_in[18];
  const float* hbias  = (const float*)d_in[19];
  float* outp = (float*)d_out;

  char* cur = (char*)d_ws;
  auto take = [&](long long bytes) -> char* {
    char* r = cur; cur += (bytes + 255) & ~(long long)255; return r;
  };

  _Float16* pwt = (_Float16*)take(2ll * D * PD);
  _Float16* wqt = (_Float16*)take(2ll * L * D * D);
  _Float16* wkt = (_Float16*)take(2ll * L * D * D);
  _Float16* wvt = (_Float16*)take(2ll * L * D * D);
  _Float16* w1t = (_Float16*)take(2ll * L * F * D);
  _Float16* w2t = (_Float16*)take(2ll * L * D * F);
  _Float16* hwt = (_Float16*)take(2ll * NCP * D);
  float*    X   = (float*)take(4ll * BS * D);
  _Float16* Yln = (_Float16*)take(2ll * BS * D);
  _Float16* Qh  = (_Float16*)take(2ll * BS * D);
  _Float16* Kh  = (_Float16*)take(2ll * BS * D);
  _Float16* Vt  = (_Float16*)take(2ll * B * H * DH * SP);
  _Float16* Pr  = (_Float16*)take(2ll * (long long)B * H * S * SP);
  float*    ctx = (float*)take(4ll * BS * D);
  _Float16* Y1  = (_Float16*)take(2ll * (long long)BS * F);
  // Aliases (only used before the transformer layers touch Pr / Y1):
  _Float16* Apat = (_Float16*)Pr;   // 19.3 MB <= 33.9 MB
  float*    PE   = (float*)Y1;      // 19.3 MB <= 38.7 MB

  auto g1 = [](long long n) { return dim3((unsigned)((n + 255) / 256)); };
  auto gg = [](int M, int N, int Z) {
    return dim3((unsigned)((N + 63) / 64), (unsigned)(((M + 31) / 32 + 7) / 8), (unsigned)Z);
  };

  // ---- weight conversion (f32 KxN -> f16 NxK, padded) ----
  { long long n = (long long)D * PD;  wt_convert<<<g1(n), 256, 0, stream>>>(patchw, pwt, PD, D, n); }
  for (int l = 0; l < L; ++l) {
    long long n = (long long)D * D;
    wt_convert<<<g1(n), 256, 0, stream>>>(wq + (long long)l * D * D, wqt + (long long)l * D * D, D, D, n);
    wt_convert<<<g1(n), 256, 0, stream>>>(wk + (long long)l * D * D, wkt + (long long)l * D * D, D, D, n);
    wt_convert<<<g1(n), 256, 0, stream>>>(wv + (long long)l * D * D, wvt + (long long)l * D * D, D, D, n);
    long long n1 = (long long)F * D;
    wt_convert<<<g1(n1), 256, 0, stream>>>(w1 + (long long)l * D * F, w1t + (long long)l * F * D, D, F, n1);
    wt_convert<<<g1(n1), 256, 0, stream>>>(w2 + (long long)l * F * D, w2t + (long long)l * D * F, F, D, n1);
  }
  { long long n = (long long)NCP * D; wt_convert<<<g1(n), 256, 0, stream>>>(hw, hwt, D, NC, n); }
  { long long n = (long long)B * H * DH * SP; zero_f16<<<g1(n), 256, 0, stream>>>(Vt, n); }

  // ---- patch embedding ----
  { long long n = (long long)BNP * PD; patchify<<<g1(n), 256, 0, stream>>>(img, Apat, n); }
  gemm_wmma<M_F32><<<gg(BNP, D, 1), 256, 0, stream>>>(
      Apat, 0, PD, pwt, 0, PD, patchb, PE, 0, D, nullptr, BNP, D, PD, 1.0f);
  { long long n = (long long)BS * D; add_cls_pos<<<g1(n), 256, 0, stream>>>(PE, cls, pos, X, n); }

  // ---- transformer layers ----
  for (int l = 0; l < L; ++l) {
    layernorm_f16<<<BS, 128, 0, stream>>>(X, D, ln1g + (long long)l * D, ln1b + (long long)l * D, Yln, D);
    gemm_wmma<M_QKH><<<gg(BS, D, 1), 256, 0, stream>>>(
        Yln, 0, D, wqt + (long long)l * D * D, 0, D, nullptr, Qh, 0, 0, nullptr, BS, D, D, 1.0f);
    gemm_wmma<M_QKH><<<gg(BS, D, 1), 256, 0, stream>>>(
        Yln, 0, D, wkt + (long long)l * D * D, 0, D, nullptr, Kh, 0, 0, nullptr, BS, D, D, 1.0f);
    gemm_wmma<M_VT><<<gg(BS, D, 1), 256, 0, stream>>>(
        Yln, 0, D, wvt + (long long)l * D * D, 0, D, nullptr, Vt, 0, 0, nullptr, BS, D, D, 1.0f);
    // scores = Q @ K^T  (batched over B*H), f16 logits into padded-width buffer
    gemm_wmma<M_F16><<<gg(S, S, B * H), 256, 0, stream>>>(
        Qh, (long long)S * DH, DH, Kh, (long long)S * DH, DH, nullptr,
        Pr, (long long)S * SP, SP, nullptr, S, S, DH, 1.0f);
    softmax_rows<<<B * H * S, 256, 0, stream>>>(Pr, 1.0f / (float)DH);
    // ctx = P @ V  (K padded to SP=224; pad probs are zero)
    gemm_wmma<M_F32><<<gg(S, DH, B * H), 256, 0, stream>>>(
        Pr, (long long)S * SP, SP, Vt, (long long)DH * SP, SP, nullptr,
        ctx, (long long)S * DH, DH, nullptr, S, DH, SP, 1.0f);
    { long long n = (long long)BS * D; add_ctx<<<g1(n), 256, 0, stream>>>(X, ctx, n); }
    layernorm_f16<<<BS, 128, 0, stream>>>(X, D, ln2g + (long long)l * D, ln2b + (long long)l * D, Yln, D);
    gemm_wmma<M_GELU_F16><<<gg(BS, F, 1), 256, 0, stream>>>(
        Yln, 0, D, w1t + (long long)l * F * D, 0, D, b1 + (long long)l * F,
        Y1, 0, F, nullptr, BS, F, D, 1.0f);
    gemm_wmma<M_RES_F32><<<gg(BS, D, 1), 256, 0, stream>>>(
        Y1, 0, F, w2t + (long long)l * D * F, 0, F, b2 + (long long)l * D,
        X, 0, D, X, BS, D, F, 1.0f);
  }

  // ---- classification head ----
  layernorm_f16<<<B, 128, 0, stream>>>(X, (long long)S * D, hg, hb, Yln, D);
  gemm_wmma<M_SCALE_F32><<<gg(B, NC, 1), 256, 0, stream>>>(
      Yln, 0, D, hwt, 0, D, hbias, outp, 0, NC, nullptr, B, NC, D,
      1.0f / sqrtf((float)D));
}
